// UniTransformerO2TwoUpdateGeneral_87548613362086
// MI455X (gfx1250) — matrix operand order repaired and checked by reading, not compiled, not run on hardware
//
#include <hip/hip_runtime.h>
#include <string.h>

typedef __attribute__((ext_vector_type(16))) _Float16 v16h;
typedef __attribute__((ext_vector_type(8)))  _Float16 v8h;
typedef __attribute__((ext_vector_type(8)))  float    v8f;
typedef _Float16 h16;

#define N_NODES 20000
#define N_EDGES 320000
#define DIM     128
#define NHEAD   16
#define GNUM    20
#define RF      80
#define KV_PAD  352      // 340 padded to 11 * 32
#define KV_NKT  11
#define KV_STRIDE  360   // halves; 720B rows -> 16B aligned, bank-staggered
#define HID_STRIDE 132   // floats
#define ACT_STRIDE 136   // halves
#define IN2_STRIDE 264   // halves (K=256 + pad)

// ---------------------------------------------------------------------------
// WMMA helpers (gfx1250, wave32): D = A(16xK f16) * B(Kx16 f16) + C(f32)
// ---------------------------------------------------------------------------
__device__ __forceinline__ v8f wmma_f16(v16h a, v16h b, v8f c) {
  return __builtin_amdgcn_wmma_f32_16x16x32_f16(false, a, false, b, (short)0, c, false, false);
}

// A fragment per ISA 7.12.2 (16-bit A 16x32): lane<16 holds K 0..7 & 16..23,
// lane>=16 holds K 8..15 & 24..31, row = lane&15. Two 16B LDS loads.
__device__ __forceinline__ v16h load_a_frag(const h16* lds, int stride, int kt, int lane) {
  int m  = lane & 15;
  int o1 = (lane & 16) ? 8 : 0;
  const h16* base = lds + m * stride + kt * 32;
  v8h lo = *(const v8h*)(base + o1);
  v8h hi = *(const v8h*)(base + o1 + 16);
  v16h a;
#pragma unroll
  for (int i = 0; i < 8; ++i) { a[i] = lo[i]; a[i + 8] = hi[i]; }
  return a;
}

// Shared tile engine: X(16xK) @ W1 -> +b1 -> LayerNorm -> ReLU -> @ W2 -> +b2.
// Block = 256 threads (8 waves); wave nt owns output columns [16nt,16nt+16).
// W1/W2 packed as B-fragments: frag (nt,kt) at ((nt*nkt+kt)*32+lane)*16 halves.
__device__ v8f mlp_tile(const h16* ldsA, int strideA, int nkt1,
                        const h16* w1p, const float* b1, const float* g1, const float* be1,
                        const h16* w2p, const float* b2, int nout2_tiles,
                        float* ldsHid, h16* ldsAct, int nt, int lane, int tid) {
  v8f acc = {};
  for (int kt = 0; kt < nkt1; ++kt) {
    v16h a = load_a_frag(ldsA, strideA, kt, lane);
    v16h b = *(const v16h*)(w1p + ((size_t)(nt * nkt1 + kt) * 32 + lane) * 16);
    acc = wmma_f16(a, b, acc);
  }
  int col   = nt * 16 + (lane & 15);
  int rbase = (lane & 16) ? 8 : 0;
  float bb = b1[col];
#pragma unroll
  for (int r = 0; r < 8; ++r) ldsHid[(rbase + r) * HID_STRIDE + col] = acc[r] + bb;
  __syncthreads();
  {
    // LayerNorm + ReLU: 16 lanes per row, shuffle-reduce within 16-lane groups
    int row = tid >> 4, j = tid & 15;
    const float* hr = ldsHid + row * HID_STRIDE;
    float s = 0.f;
    for (int c = j; c < DIM; c += 16) s += hr[c];
#pragma unroll
    for (int m = 1; m < 16; m <<= 1) s += __shfl_xor(s, m, 32);
    float mu = s * (1.0f / DIM);
    float v = 0.f;
    for (int c = j; c < DIM; c += 16) { float d = hr[c] - mu; v += d * d; }
#pragma unroll
    for (int m = 1; m < 16; m <<= 1) v += __shfl_xor(v, m, 32);
    float inv = rsqrtf(v * (1.0f / DIM) + 1e-5f);
    for (int c = j; c < DIM; c += 16) {
      float y = (hr[c] - mu) * inv * g1[c] + be1[c];
      ldsAct[row * ACT_STRIDE + c] = (h16)(y > 0.f ? y : 0.f);
    }
  }
  __syncthreads();
  v8f acc2 = {};
  if (nt < nout2_tiles) {          // wave-uniform: EXEC stays all-ones for WMMA
    for (int kt = 0; kt < 4; ++kt) {
      v16h a = load_a_frag(ldsAct, ACT_STRIDE, kt, lane);
      v16h b = *(const v16h*)(w2p + ((size_t)(nt * 4 + kt) * 32 + lane) * 16);
      acc2 = wmma_f16(a, b, acc2);
    }
    float b2v = b2[col];
#pragma unroll
    for (int r = 0; r < 8; ++r) acc2[r] += b2v;
  }
  return acc2;
}

// ---------------------------------------------------------------------------
// Weight packing: W row-major [K][N] (f32) -> f16 B-fragments
// ---------------------------------------------------------------------------
__global__ void pack_b_kernel(const float* W, int K, int N, int nKT, h16* out) {
  int gid = blockIdx.x * blockDim.x + threadIdx.x;
  int total = nKT * (N / 16) * 32;
  if (gid >= total) return;
  int lane = gid & 31, tile = gid >> 5;
  int nt = tile / nKT, kt = tile % nKT;
  int n = nt * 16 + (lane & 15);
  int kbase = kt * 32 + ((lane & 16) ? 8 : 0);
  h16* dst = out + (size_t)gid * 16;
#pragma unroll
  for (int j = 0; j < 16; ++j) {
    int k = kbase + (j < 8 ? j : j + 8);
    dst[j] = (h16)((k < K) ? W[(size_t)k * N + n] : 0.0f);
  }
}

// ---------------------------------------------------------------------------
// Per-edge geometry: rel_x, Gaussian r_feat (f16), both sigmoid edge gates
// ---------------------------------------------------------------------------
__global__ void edge_geom_kernel(const float* x, const float* eattr, const int* ei,
                                 const float* eww1, const float* ewb1,
                                 const float* eww2, const float* ewb2,
                                 float* relx, h16* rfeat, float* ew1, float* ew2) {
  int e = blockIdx.x * blockDim.x + threadIdx.x;
  if (e >= N_EDGES) return;
  int s = ei[e], d = ei[N_EDGES + e];
  float rx = x[d * 3 + 0] - x[s * 3 + 0];
  float ry = x[d * 3 + 1] - x[s * 3 + 1];
  float rz = x[d * 3 + 2] - x[s * 3 + 2];
  relx[e * 3 + 0] = rx; relx[e * 3 + 1] = ry; relx[e * 3 + 2] = rz;
  float dist = sqrtf(rx * rx + ry * ry + rz * rz);
  float ga0 = eattr[(size_t)e * 4 + 0], ga1 = eattr[(size_t)e * 4 + 1];
  float ga2 = eattr[(size_t)e * 4 + 2], ga3 = eattr[(size_t)e * 4 + 3];
  const float step  = 10.0f / 19.0f;
  const float coeff = -0.5f / (step * step);
  float z1 = ewb1[0], z2 = ewb2[0];
  h16* rf = rfeat + (size_t)e * RF;
  for (int g = 0; g < GNUM; ++g) {
    float dd = dist - (float)g * step;
    float gv = __expf(coeff * dd * dd);
    float r0 = ga0 * gv, r1 = ga1 * gv, r2 = ga2 * gv, r3 = ga3 * gv;
    rf[0 * GNUM + g] = (h16)r0; rf[1 * GNUM + g] = (h16)r1;
    rf[2 * GNUM + g] = (h16)r2; rf[3 * GNUM + g] = (h16)r3;
    z1 += r0 * eww1[0 * GNUM + g] + r1 * eww1[1 * GNUM + g] + r2 * eww1[2 * GNUM + g] + r3 * eww1[3 * GNUM + g];
    z2 += r0 * eww2[0 * GNUM + g] + r1 * eww2[1 * GNUM + g] + r2 * eww2[2 * GNUM + g] + r3 * eww2[3 * GNUM + g];
  }
  ew1[e] = 1.f / (1.f + __expf(-z1));
  ew2[e] = 1.f / (1.f + __expf(-z2));
}

__global__ void fill_u32_kernel(unsigned* p, long n, unsigned v) {
  long i = (long)blockIdx.x * blockDim.x + threadIdx.x;
  long st = (long)gridDim.x * blockDim.x;
  for (; i < n; i += st) p[i] = v;
}

// ---------------------------------------------------------------------------
// Fused per-edge-tile attention kernel: kv build -> K MLP -> scores/segmax
//                                              -> V MLP -> gated v store
// v_ntiles==8 : x2h (v is [E,128] f16);  v_ntiles==1 : h2x (v is [E,16] f32)
// ---------------------------------------------------------------------------
__global__ __launch_bounds__(256) void edge_attn_kernel(
    const float* h_in, const float* eattr, const int* ei,
    const h16* rfeat, const float* ew, const float* qbuf,
    const h16* wk1, const float* bk1, const float* gk1, const float* bek1,
    const h16* wk2, const float* bk2,
    const h16* wv1, const float* bv1, const float* gv1, const float* bev1,
    const h16* wv2, const float* bv2, int v_ntiles,
    float* scores, unsigned* maxenc, h16* v16out, float* v32out) {
  __shared__ __align__(16) h16   s_kv[16 * KV_STRIDE];
  __shared__ __align__(16) float s_hid[16 * HID_STRIDE];
  __shared__ __align__(16) h16   s_act[16 * ACT_STRIDE];
  __shared__ int s_dst[16], s_src[16];
  __shared__ float s_ew[16];
  int tid = threadIdx.x, lane = tid & 31, nt = tid >> 5;
  int e0 = blockIdx.x * 16;
  if (tid < 16) {
    s_src[tid] = ei[e0 + tid];
    s_dst[tid] = ei[N_EDGES + e0 + tid];
    s_ew[tid]  = ew[e0 + tid];
  }
  __syncthreads();
  for (int idx = tid; idx < 16 * KV_PAD; idx += 256) {
    int r = idx / KV_PAD, c = idx - r * KV_PAD;
    int e = e0 + r;
    float v;
    if (c < 4)        v = eattr[(size_t)e * 4 + c];
    else if (c < 84)  v = (float)rfeat[(size_t)e * RF + (c - 4)];
    else if (c < 212) v = h_in[(size_t)s_dst[r] * DIM + (c - 84)];
    else if (c < 340) v = h_in[(size_t)s_src[r] * DIM + (c - 212)];
    else              v = 0.f;
    s_kv[r * KV_STRIDE + c] = (h16)v;
  }
  __syncthreads();

  // ---- K branch: scores = (q[dst] * k).sum(head) / sqrt(8), segment max ----
  v8f kk = mlp_tile(s_kv, KV_STRIDE, KV_NKT, wk1, bk1, gk1, bek1, wk2, bk2, 8,
                    s_hid, s_act, nt, lane, tid);
  {
    int col = nt * 16 + (lane & 15);
    int rbase = (lane & 16) ? 8 : 0;
    int head = col >> 3;
#pragma unroll
    for (int r = 0; r < 8; ++r) {
      int row = rbase + r;
      int d = s_dst[row];
      float val = kk[r] * qbuf[(size_t)d * DIM + col];
#pragma unroll
      for (int m = 1; m < 8; m <<= 1) val += __shfl_xor(val, m, 32);
      if ((lane & 7) == 0) {
        float sc = val * 0.35355339059327f;   // 1/sqrt(8)
        scores[(size_t)(e0 + row) * NHEAD + head] = sc;
        unsigned u = __float_as_uint(sc);
        unsigned enc = (u & 0x80000000u) ? ~u : (u | 0x80000000u);
        atomicMax(&maxenc[(size_t)d * NHEAD + head], enc);
      }
    }
  }

  // ---- V branch: gated values ----
  v8f vv = mlp_tile(s_kv, KV_STRIDE, KV_NKT, wv1, bv1, gv1, bev1, wv2, bv2, v_ntiles,
                    s_hid, s_act, nt, lane, tid);
  if (nt < v_ntiles) {
    int col = nt * 16 + (lane & 15);
    int rbase = (lane & 16) ? 8 : 0;
    if (v_ntiles == 8) {
#pragma unroll
      for (int r = 0; r < 8; ++r) {
        int row = rbase + r;
        v16out[(size_t)(e0 + row) * DIM + col] = (h16)(vv[r] * s_ew[row]);
      }
    } else {
#pragma unroll
      for (int r = 0; r < 8; ++r) {
        int row = rbase + r;
        v32out[(size_t)(e0 + row) * NHEAD + col] = vv[r] * s_ew[row];
      }
    }
  }
}

// ---------------------------------------------------------------------------
// Node MLP (128 -> 128 -> 128): q projections
// ---------------------------------------------------------------------------
__global__ __launch_bounds__(256) void node_mlp_kernel(
    const float* X, const h16* w1p, const float* b1, const float* g1, const float* be1,
    const h16* w2p, const float* b2, float* out) {
  __shared__ __align__(16) h16   s_in[16 * ACT_STRIDE];
  __shared__ __align__(16) float s_hid[16 * HID_STRIDE];
  __shared__ __align__(16) h16   s_act[16 * ACT_STRIDE];
  int tid = threadIdx.x, lane = tid & 31, nt = tid >> 5;
  int n0 = blockIdx.x * 16;
  for (int idx = tid; idx < 16 * DIM; idx += 256) {
    int r = idx >> 7, c = idx & 127;
    s_in[r * ACT_STRIDE + c] = (h16)X[(size_t)(n0 + r) * DIM + c];
  }
  __syncthreads();
  v8f acc = mlp_tile(s_in, ACT_STRIDE, 4, w1p, b1, g1, be1, w2p, b2, 8,
                     s_hid, s_act, nt, lane, tid);
  int col = nt * 16 + (lane & 15);
  int rbase = (lane & 16) ? 8 : 0;
#pragma unroll
  for (int r = 0; r < 8; ++r) out[(size_t)(n0 + rbase + r) * DIM + col] = acc[r];
}

// ---------------------------------------------------------------------------
// Output MLP: concat(attn_out, h) (256) -> 128 -> 128, + residual h
// ---------------------------------------------------------------------------
__global__ __launch_bounds__(256) void out_mlp_kernel(
    const float* accum, const float* h_in,
    const h16* w1p, const float* b1, const float* g1, const float* be1,
    const h16* w2p, const float* b2, float* out) {
  __shared__ __align__(16) h16   s_in[16 * IN2_STRIDE];
  __shared__ __align__(16) float s_hid[16 * HID_STRIDE];
  __shared__ __align__(16) h16   s_act[16 * ACT_STRIDE];
  int tid = threadIdx.x, lane = tid & 31, nt = tid >> 5;
  int n0 = blockIdx.x * 16;
  for (int idx = tid; idx < 16 * 256; idx += 256) {
    int r = idx >> 8, c = idx & 255;
    float v = (c < 128) ? accum[(size_t)(n0 + r) * DIM + c]
                        : h_in[(size_t)(n0 + r) * DIM + (c - 128)];
    s_in[r * IN2_STRIDE + c] = (h16)v;
  }
  __syncthreads();
  v8f acc = mlp_tile(s_in, IN2_STRIDE, 8, w1p, b1, g1, be1, w2p, b2, 8,
                     s_hid, s_act, nt, lane, tid);
  int col = nt * 16 + (lane & 15);
  int rbase = (lane & 16) ? 8 : 0;
#pragma unroll
  for (int r = 0; r < 8; ++r) {
    size_t o = (size_t)(n0 + rbase + r) * DIM + col;
    out[o] = acc[r] + h_in[o];
  }
}

// ---------------------------------------------------------------------------
// Scatter softmax passes
// ---------------------------------------------------------------------------
__device__ __forceinline__ float dec_ord(unsigned u) {
  unsigned r = (u & 0x80000000u) ? (u & 0x7fffffffu) : ~u;
  return __uint_as_float(r);
}

__global__ void softmax_exp_kernel(const int* ei, float* sc, const unsigned* maxenc,
                                   float* denom) {
  long gid = (long)blockIdx.x * blockDim.x + threadIdx.x;
  if (gid >= (long)N_EDGES * NHEAD) return;
  int e = (int)(gid >> 4), hd = (int)(gid & 15);
  int d = ei[N_EDGES + e];
  float mx = dec_ord(maxenc[(size_t)d * NHEAD + hd]);
  float ex = __expf(sc[gid] - mx);
  sc[gid] = ex;
  atomicAdd(&denom[(size_t)d * NHEAD + hd], ex);
}

__global__ void aggregate_h_kernel(const int* ei, const float* ex, const float* denom,
                                   const h16* v16, float* acc) {
  long gid = (long)blockIdx.x * blockDim.x + threadIdx.x;
  if (gid >= (long)N_EDGES * DIM) return;
  int e = (int)(gid >> 7), c = (int)(gid & 127), hd = c >> 3;
  int d = ei[N_EDGES + e];
  float alpha = ex[(size_t)e * NHEAD + hd] / (denom[(size_t)d * NHEAD + hd] + 1e-16f);
  atomicAdd(&acc[(size_t)d * DIM + c], alpha * (float)v16[gid]);
}

__global__ void aggregate_x_kernel(const int* ei, const float* ex, const float* denom,
                                   const float* v2, const float* relx, float* accx) {
  int e = blockIdx.x * blockDim.x + threadIdx.x;
  if (e >= N_EDGES) return;
  int d = ei[N_EDGES + e];
  float s = 0.f;
#pragma unroll
  for (int hd = 0; hd < NHEAD; ++hd)
    s += ex[(size_t)e * NHEAD + hd] / (denom[(size_t)d * NHEAD + hd] + 1e-16f) *
         v2[(size_t)e * NHEAD + hd];
  atomicAdd(&accx[d * 3 + 0], s * relx[e * 3 + 0]);
  atomicAdd(&accx[d * 3 + 1], s * relx[e * 3 + 1]);
  atomicAdd(&accx[d * 3 + 2], s * relx[e * 3 + 2]);
}

__global__ void finalize_x_kernel(const float* x, const float* accx, float* out) {
  int i = blockIdx.x * blockDim.x + threadIdx.x;
  if (i >= N_NODES * 3) return;
  out[i] = x[i] + accx[i] * (1.0f / NHEAD);
}

// ---------------------------------------------------------------------------
// Host launcher
// ---------------------------------------------------------------------------
// Input index map (insertion-order flatten of setup_inputs):
// 0 h, 1 x, 2 edge_attr, 3 edge_index, 4 mask_ligand (unused),
// params: x2h{hk,hv,hq,out: each w1,b1,g1,beta1,w2,b2; ew_w, ew_b}, then
//         h2x{xk,xv,xq: same; ew_w, ew_b}
#define PF(i) ((const float*)d_in[i])

extern "C" void kernel_launch(void* const* d_in, const int* in_sizes, int n_in,
                              void* d_out, int out_size, void* d_ws, size_t ws_size,
                              hipStream_t stream) {
  (void)in_sizes; (void)n_in; (void)out_size;
  const float* h_p  = PF(0);
  const float* x_p  = PF(1);
  const float* ea_p = PF(2);
  const int*   ei_p = (const int*)d_in[3];

  const int HK = 5, HV = 11, HQ = 17, OUTP = 23;        // x2h mlps
  const int X2H_EWW = 29, X2H_EWB = 30;
  const int XK = 31, XV = 37, XQ = 43;                  // h2x mlps
  const int H2X_EWW = 49, H2X_EWB = 50;
  // per-mlp leaf offsets: +0 w1, +1 b1, +2 g1, +3 beta1, +4 w2, +5 b2

  char* ws = (char*)d_ws;
  size_t off = 0;
  auto alloc = [&](size_t bytes) -> void* {
    void* p = ws + off;
    off += (bytes + 255) & ~(size_t)255;
    return p;
  };
  // packed weights (f16 B-fragments): nKT * (N/16) * 512 halves
  h16* w_hk1 = (h16*)alloc(11 * 8 * 512 * 2);  h16* w_hk2 = (h16*)alloc(4 * 8 * 512 * 2);
  h16* w_hv1 = (h16*)alloc(11 * 8 * 512 * 2);  h16* w_hv2 = (h16*)alloc(4 * 8 * 512 * 2);
  h16* w_hq1 = (h16*)alloc(4 * 8 * 512 * 2);   h16* w_hq2 = (h16*)alloc(4 * 8 * 512 * 2);
  h16* w_ot1 = (h16*)alloc(8 * 8 * 512 * 2);   h16* w_ot2 = (h16*)alloc(4 * 8 * 512 * 2);
  h16* w_xk1 = (h16*)alloc(11 * 8 * 512 * 2);  h16* w_xk2 = (h16*)alloc(4 * 8 * 512 * 2);
  h16* w_xv1 = (h16*)alloc(11 * 8 * 512 * 2);  h16* w_xv2 = (h16*)alloc(4 * 1 * 512 * 2);
  h16* w_xq1 = (h16*)alloc(4 * 8 * 512 * 2);   h16* w_xq2 = (h16*)alloc(4 * 8 * 512 * 2);
  // scratch buffers
  h16*   rfeat  = (h16*)alloc((size_t)N_EDGES * RF * 2);
  float* ew1    = (float*)alloc((size_t)N_EDGES * 4);
  float* ew2    = (float*)alloc((size_t)N_EDGES * 4);
  float* relx   = (float*)alloc((size_t)N_EDGES * 3 * 4);
  float* qbuf   = (float*)alloc((size_t)N_NODES * DIM * 4);
  float* scores = (float*)alloc((size_t)N_EDGES * NHEAD * 4);
  h16*   v16b   = (h16*)alloc((size_t)N_EDGES * DIM * 2);
  float* v2b    = (float*)alloc((size_t)N_EDGES * NHEAD * 4);
  unsigned* mxe = (unsigned*)alloc((size_t)N_NODES * NHEAD * 4);
  float* denom  = (float*)alloc((size_t)N_NODES * NHEAD * 4);
  float* acc_h  = (float*)alloc((size_t)N_NODES * DIM * 4);
  float* acc_x  = (float*)alloc((size_t)N_NODES * 3 * 4);
  if (off > ws_size) return;   // workspace too small; deterministic no-op

  float* out_h = (float*)d_out;                 // [N,128] x2h_out (also new_h)
  float* out_x = (float*)d_out + (size_t)N_NODES * DIM;  // [N,3] x_new

  auto pack = [&](const float* W, int K, int N, int nKT, h16* dst) {
    int total = nKT * (N / 16) * 32;
    pack_b_kernel<<<(total + 255) / 256, 256, 0, stream>>>(W, K, N, nKT, dst);
  };
  pack(PF(HK + 0), 340, 128, 11, w_hk1);  pack(PF(HK + 4), 128, 128, 4, w_hk2);
  pack(PF(HV + 0), 340, 128, 11, w_hv1);  pack(PF(HV + 4), 128, 128, 4, w_hv2);
  pack(PF(HQ + 0), 128, 128, 4, w_hq1);   pack(PF(HQ + 4), 128, 128, 4, w_hq2);
  pack(PF(OUTP + 0), 256, 128, 8, w_ot1); pack(PF(OUTP + 4), 128, 128, 4, w_ot2);
  pack(PF(XK + 0), 340, 128, 11, w_xk1);  pack(PF(XK + 4), 128, 128, 4, w_xk2);
  pack(PF(XV + 0), 340, 128, 11, w_xv1);  pack(PF(XV + 4), 128, 16, 4, w_xv2);
  pack(PF(XQ + 0), 128, 128, 4, w_xq1);   pack(PF(XQ + 4), 128, 128, 4, w_xq2);

  edge_geom_kernel<<<(N_EDGES + 255) / 256, 256, 0, stream>>>(
      x_p, ea_p, ei_p, PF(X2H_EWW), PF(X2H_EWB), PF(H2X_EWW), PF(H2X_EWB),
      relx, rfeat, ew1, ew2);

  float ninf = -3.0e38f; unsigned um; memcpy(&um, &ninf, 4);
  unsigned enc_ninf = ~um;   // order-preserving encoding of a negative float

  // ---- phase 1: x2h ----
  fill_u32_kernel<<<2048, 256, 0, stream>>>(mxe, (long)N_NODES * NHEAD, enc_ninf);
  fill_u32_kernel<<<2048, 256, 0, stream>>>((unsigned*)denom, (long)N_NODES * NHEAD, 0u);
  fill_u32_kernel<<<2048, 256, 0, stream>>>((unsigned*)acc_h, (long)N_NODES * DIM, 0u);
  fill_u32_kernel<<<2048, 256, 0, stream>>>((unsigned*)acc_x, (long)N_NODES * 3, 0u);

  node_mlp_kernel<<<N_NODES / 16, 256, 0, stream>>>(
      h_p, w_hq1, PF(HQ + 1), PF(HQ + 2), PF(HQ + 3), w_hq2, PF(HQ + 5), qbuf);

  edge_attn_kernel<<<N_EDGES / 16, 256, 0, stream>>>(
      h_p, ea_p, ei_p, rfeat, ew1, qbuf,
      w_hk1, PF(HK + 1), PF(HK + 2), PF(HK + 3), w_hk2, PF(HK + 5),
      w_hv1, PF(HV + 1), PF(HV + 2), PF(HV + 3), w_hv2, PF(HV + 5), 8,
      scores, mxe, v16b, nullptr);

  softmax_exp_kernel<<<(int)(((long)N_EDGES * NHEAD + 255) / 256), 256, 0, stream>>>(
      ei_p, scores, mxe, denom);
  aggregate_h_kernel<<<(int)(((long)N_EDGES * DIM + 255) / 256), 256, 0, stream>>>(
      ei_p, scores, denom, v16b, acc_h);

  out_mlp_kernel<<<N_NODES / 16, 256, 0, stream>>>(
      acc_h, h_p, w_ot1, PF(OUTP + 1), PF(OUTP + 2), PF(OUTP + 3),
      w_ot2, PF(OUTP + 5), out_h);

  // ---- phase 2: h2x (new_h = out_h) ----
  fill_u32_kernel<<<2048, 256, 0, stream>>>(mxe, (long)N_NODES * NHEAD, enc_ninf);
  fill_u32_kernel<<<2048, 256, 0, stream>>>((unsigned*)denom, (long)N_NODES * NHEAD, 0u);

  node_mlp_kernel<<<N_NODES / 16, 256, 0, stream>>>(
      out_h, w_xq1, PF(XQ + 1), PF(XQ + 2), PF(XQ + 3), w_xq2, PF(XQ + 5), qbuf);

  edge_attn_kernel<<<N_EDGES / 16, 256, 0, stream>>>(
      out_h, ea_p, ei_p, rfeat, ew2, qbuf,
      w_xk1, PF(XK + 1), PF(XK + 2), PF(XK + 3), w_xk2, PF(XK + 5),
      w_xv1, PF(XV + 1), PF(XV + 2), PF(XV + 3), w_xv2, PF(XV + 5), 1,
      scores, mxe, nullptr, v2b);

  softmax_exp_kernel<<<(int)(((long)N_EDGES * NHEAD + 255) / 256), 256, 0, stream>>>(
      ei_p, scores, mxe, denom);
  aggregate_x_kernel<<<(N_EDGES + 255) / 256, 256, 0, stream>>>(
      ei_p, scores, denom, v2b, relx, acc_x);

  finalize_x_kernel<<<(N_NODES * 3 + 255) / 256, 256, 0, stream>>>(x_p, acc_x, out_x);
}
